// GraphMambaSpatialV3_63952063037505
// MI455X (gfx1250) — matrix operand order repaired and checked by reading, not compiled, not run on hardware
//
#include <hip/hip_runtime.h>
#include <hip/hip_bf16.h>
#include <math.h>

#define B_    128
#define N_    47
#define DIN_  64
#define D_    896
#define H_    14
#define L_    12
#define DH_   64
#define DHID_ 1792
#define TOK_  (B_*N_)

typedef __attribute__((ext_vector_type(4)))  __bf16 v4bf;
typedef __attribute__((ext_vector_type(8)))  __bf16 v8bf;
typedef __attribute__((ext_vector_type(16))) __bf16 v16bf;
typedef __attribute__((ext_vector_type(8)))  float  v8f;

// Epilogue bits
#define E_BIAS    1
#define E_SILU    2
#define E_RESID   4
#define E_STATION 8
#define E_OUTF    16
#define E_OUTB    32

// ---------------------------------------------------------------------------
// C[M,N] = epilogue( A[M,K](bf16) @ W[K,N](fp32->bf16) )
// Block: 128 threads (4 waves). Tile: 128(M) x 64(N) x 32(K), double-buffered.
// Wave w computes rows [w*32, w*32+32) x 64 cols = 8 WMMAs per K-step.
// A tile: CDNA5 async copy global->LDS (bf16, padded stride via per-lane addrs).
// W tile: fp32 global -> regs -> bf16 packed LDS stores (K-transposed).
// ---------------------------------------------------------------------------
template<int EPI>
__global__ __launch_bounds__(128) void gemm_bf16_wmma(
    const __bf16* __restrict__ A, const float* __restrict__ W,
    const float* __restrict__ bias, const float* __restrict__ residual,
    const float* __restrict__ station,
    float* __restrict__ Cf, __bf16* __restrict__ Cb,
    int M, int Nn, int K)
{
  constexpr int BM = 128, BN = 64, BK = 32, AST = 40; // padded bf16 stride
  __shared__ __attribute__((aligned(16))) __bf16 As[2][BM * AST];
  __shared__ __attribute__((aligned(16))) __bf16 Bs[2][BN * AST];

  const int tid  = threadIdx.x;
  const int wave = tid >> 5;
  const int lane = tid & 31;
  const int m0 = blockIdx.y * BM;
  const int n0 = blockIdx.x * BN;
  const int mm = lane & 15;
  const int hi = lane >> 4;

  v8f acc[2][4];
  #pragma unroll
  for (int ai = 0; ai < 2; ++ai)
    #pragma unroll
    for (int j = 0; j < 4; ++j)
      #pragma unroll
      for (int i = 0; i < 8; ++i) acc[ai][j][i] = 0.0f;

  // ---- async A tile copy: 128 rows x 64B, 4x16B chunks per thread ----
  auto issueA = [&](int k0, int buf) {
    #pragma unroll
    for (int i = 0; i < 4; ++i) {
      int lin = tid + i * 128;
      int row = lin >> 2;
      int seg = lin & 3;
      const __bf16* gp = A + (size_t)(m0 + row) * K + k0 + seg * 8;
      unsigned lds = (unsigned)(unsigned long long)(&As[buf][row * AST + seg * 8]);
      asm volatile("global_load_async_to_lds_b128 %0, %1, off"
                   :: "v"(lds), "v"(gp) : "memory");
    }
  };

  // ---- W tile: 32x64 fp32, 16 scalars per thread (coalesced across n) ----
  float breg[16];
  auto loadB = [&](int k0) {
    #pragma unroll
    for (int i = 0; i < 2; ++i) {
      int lin = tid + i * 128;
      int nn = lin & 63;
      int sg = lin >> 6;          // 0..3 : K sub-segment of 8
      #pragma unroll
      for (int j = 0; j < 8; ++j)
        breg[i * 8 + j] = W[(size_t)(k0 + sg * 8 + j) * Nn + n0 + nn];
    }
  };
  auto storeB = [&](int buf) {
    #pragma unroll
    for (int i = 0; i < 2; ++i) {
      int lin = tid + i * 128;
      int nn = lin & 63;
      int sg = lin >> 6;
      v8bf p;
      #pragma unroll
      for (int j = 0; j < 8; ++j) p[j] = (__bf16)breg[i * 8 + j];
      *(v8bf*)(&Bs[buf][nn * AST + sg * 8]) = p;   // Bs[n][k], 16B aligned
    }
  };

  auto compute = [&](int buf) {
    // 4 B fragments (32x16 each): lane = col mm, K = [hi*8..hi*8+7 | 16+hi*8..16+hi*8+7]
    v16bf bfrag[4];
    #pragma unroll
    for (int j = 0; j < 4; ++j) {
      v8bf blo = *(const v8bf*)(&Bs[buf][(j * 16 + mm) * AST + hi * 8]);
      v8bf bhi = *(const v8bf*)(&Bs[buf][(j * 16 + mm) * AST + 16 + hi * 8]);
      #pragma unroll
      for (int i = 0; i < 8; ++i) { bfrag[j][i] = blo[i]; bfrag[j][i + 8] = bhi[i]; }
    }
    #pragma unroll
    for (int ai = 0; ai < 2; ++ai) {
      // A fragment 16x32: lanes 0-15 K=[0..7|16..23]; lanes 16-31 K=[8..15|24..31]
      int arow = wave * 32 + ai * 16 + mm;
      v8bf alo = *(const v8bf*)(&As[buf][arow * AST + hi * 8]);
      v8bf ahi = *(const v8bf*)(&As[buf][arow * AST + 16 + hi * 8]);
      v16bf af;
      #pragma unroll
      for (int i = 0; i < 8; ++i) { af[i] = alo[i]; af[i + 8] = ahi[i]; }
      #pragma unroll
      for (int j = 0; j < 4; ++j)
        acc[ai][j] = __builtin_amdgcn_wmma_f32_16x16x32_bf16(
            false, af, false, bfrag[j], (short)0, acc[ai][j], false, false);
    }
  };

  const int ktiles = K / BK;
  issueA(0, 0);
  loadB(0);
  storeB(0);
  asm volatile("s_wait_asynccnt 0x0" ::: "memory");
  __syncthreads();

  for (int kt = 0; kt < ktiles; ++kt) {
    const int buf = kt & 1;
    if (kt + 1 < ktiles) {
      issueA((kt + 1) * BK, buf ^ 1);   // overlaps with compute below
      loadB((kt + 1) * BK);
    }
    compute(buf);
    if (kt + 1 < ktiles) {
      storeB(buf ^ 1);
      asm volatile("s_wait_asynccnt 0x0" ::: "memory");
      __syncthreads();
    }
  }

  // ---- branch-free epilogue (template-selected) ----
  #pragma unroll
  for (int ai = 0; ai < 2; ++ai) {
    #pragma unroll
    for (int j = 0; j < 4; ++j) {
      int col = n0 + j * 16 + mm;
      float bv = 0.0f;
      if constexpr (EPI & E_BIAS) bv = bias[col];
      #pragma unroll
      for (int i = 0; i < 8; ++i) {
        int row = m0 + wave * 32 + ai * 16 + hi * 8 + i;
        float c = acc[ai][j][i] + bv;
        if constexpr (EPI & E_SILU) c = c / (1.0f + __expf(-c));
        size_t off = (size_t)row * Nn + col;
        if constexpr (EPI & E_RESID) c += residual[off];
        if constexpr (EPI & E_STATION) c += station[(size_t)(row % N_) * Nn + col];
        if constexpr (EPI & E_OUTF) Cf[off] = c;
        if constexpr (EPI & E_OUTB) Cb[off] = (__bf16)c;
      }
    }
  }
}

// ---------------------------------------------------------------------------
// fp32 -> bf16 convert (for the network input x)
// ---------------------------------------------------------------------------
__global__ void cvt_kernel(const float* __restrict__ in, __bf16* __restrict__ outp, int n)
{
  int i = blockIdx.x * blockDim.x + threadIdx.x;
  if (i < n) outp[i] = (__bf16)in[i];
}

// ---------------------------------------------------------------------------
// LayerNorm: one block per token (D=896); emits bf16 (consumed only by WMMA)
// ---------------------------------------------------------------------------
__global__ __launch_bounds__(256) void ln_kernel(
    const float* __restrict__ x, const float* __restrict__ g,
    const float* __restrict__ b, __bf16* __restrict__ y)
{
  const int t = blockIdx.x;
  const float* xr = x + (size_t)t * D_;
  __bf16* yr = y + (size_t)t * D_;
  __shared__ float sred[8];
  const int lane = threadIdx.x & 31, wave = threadIdx.x >> 5;

  float s = 0.0f;
  for (int i = threadIdx.x; i < D_; i += 256) s += xr[i];
  #pragma unroll
  for (int m = 16; m > 0; m >>= 1) s += __shfl_xor(s, m, 32);
  if (lane == 0) sred[wave] = s;
  __syncthreads();
  float mean = 0.0f;
  #pragma unroll
  for (int w = 0; w < 8; ++w) mean += sred[w];
  mean *= (1.0f / D_);
  __syncthreads();

  float v = 0.0f;
  for (int i = threadIdx.x; i < D_; i += 256) { float d = xr[i] - mean; v += d * d; }
  #pragma unroll
  for (int m = 16; m > 0; m >>= 1) v += __shfl_xor(v, m, 32);
  if (lane == 0) sred[wave] = v;
  __syncthreads();
  float var = 0.0f;
  #pragma unroll
  for (int w = 0; w < 8; ++w) var += sred[w];
  float rstd = rsqrtf(var * (1.0f / D_) + 1e-5f);

  for (int i = threadIdx.x; i < D_; i += 256)
    yr[i] = (__bf16)((xr[i] - mean) * rstd * g[i] + b[i]);
}

// ---------------------------------------------------------------------------
// RoPE (in place on q and k, fp32)
// ---------------------------------------------------------------------------
__global__ void rope_kernel(float* __restrict__ q, float* __restrict__ k)
{
  const int total = TOK_ * H_ * 32;
  int idx = blockIdx.x * blockDim.x + threadIdx.x;
  if (idx >= total) return;
  int i = idx & 31;
  int h = (idx >> 5) % H_;
  int t = idx / (H_ * 32);
  int n = t % N_;
  float inv_freq = __expf(-(float)i * (9.210340371976184f / 32.0f)); // 10000^(-i/32)
  float ang = (float)n * inv_freq;
  float c = __cosf(ang), s = __sinf(ang);
  size_t base = (size_t)t * D_ + h * DH_ + i;
  float q1 = q[base], q2 = q[base + 32];
  q[base] = q1 * c - q2 * s; q[base + 32] = q2 * c + q1 * s;
  float k1 = k[base], k2 = k[base + 32];
  k[base] = k1 * c - k2 * s; k[base + 32] = k2 * c + k1 * s;
}

// ---------------------------------------------------------------------------
// Attention: one block per (b,h); emits bf16 o (consumed only by Wo GEMM)
// ---------------------------------------------------------------------------
__global__ __launch_bounds__(256) void attn_kernel(
    const float* __restrict__ q, const float* __restrict__ k,
    const float* __restrict__ v, const float* __restrict__ adj,
    const float* __restrict__ bearings, const float* __restrict__ wind_dirs,
    const float* __restrict__ windW, const float* __restrict__ windB,
    __bf16* __restrict__ o)
{
  const int b = blockIdx.x / H_;
  const int h = blockIdx.x % H_;
  __shared__ float ks[N_][DH_ + 1];
  __shared__ float vs[N_][DH_ + 1];
  __shared__ float qs[8][DH_];
  __shared__ float ar[8][N_ + 1];

  for (int i = threadIdx.x; i < N_ * DH_; i += 256) {
    int n = i >> 6, d = i & 63;
    size_t off = ((size_t)(b * N_ + n)) * D_ + h * DH_ + d;
    ks[n][d] = k[off];
    vs[n][d] = v[off];
  }
  __syncthreads();

  const int wave = threadIdx.x >> 5, lane = threadIdx.x & 31;
  const float w0 = windW[h], w1 = windW[H_ + h], wb = windB[h];
  const float scale = 0.125f; // 1/sqrt(64)

  for (int n = wave; n < N_; n += 8) {
    size_t qoff = ((size_t)(b * N_ + n)) * D_ + h * DH_;
    qs[wave][lane]      = q[qoff + lane];
    qs[wave][lane + 32] = q[qoff + 32 + lane];
    __builtin_amdgcn_wave_barrier();

    float wd = wind_dirs[b * N_ + n];
    float lg[2];
    #pragma unroll
    for (int p = 0; p < 2; ++p) {
      int m = lane + p * 32;
      lg[p] = -__builtin_inff();
      if (m < N_) {
        float s = 0.0f;
        #pragma unroll 8
        for (int d = 0; d < DH_; ++d) s += qs[wave][d] * ks[m][d];
        float a = adj[n * N_ + m];
        float lmask = (a > 0.0f) ? s * scale : -__builtin_inff();
        float rev = fmodf(bearings[m * N_ + n] + 180.0f, 360.0f);
        float align = __cosf((wd - rev) * 0.017453292519943295f);
        float bias = tanhf(align * w0 + a * w1 + wb);
        lg[p] = lmask + bias;
      }
    }
    float mx = fmaxf(lg[0], lg[1]);
    #pragma unroll
    for (int m = 16; m > 0; m >>= 1) mx = fmaxf(mx, __shfl_xor(mx, m, 32));
    float e0 = (lane      < N_) ? __expf(lg[0] - mx) : 0.0f;
    float e1 = (lane + 32 < N_) ? __expf(lg[1] - mx) : 0.0f;
    float sum = e0 + e1;
    #pragma unroll
    for (int m = 16; m > 0; m >>= 1) sum += __shfl_xor(sum, m, 32);
    float inv = 1.0f / sum;
    ar[wave][lane] = e0 * inv;
    if (lane + 32 < N_) ar[wave][lane + 32] = e1 * inv;
    __builtin_amdgcn_wave_barrier();

    float o0 = 0.0f, o1 = 0.0f;
    for (int m = 0; m < N_; ++m) {
      float p = ar[wave][m];
      o0 += p * vs[m][lane];
      o1 += p * vs[m][lane + 32];
    }
    o[qoff + lane]      = (__bf16)o0;
    o[qoff + 32 + lane] = (__bf16)o1;
  }
}

// ---------------------------------------------------------------------------
extern "C" void kernel_launch(void* const* d_in, const int* in_sizes, int n_in,
                              void* d_out, int out_size, void* d_ws, size_t ws_size,
                              hipStream_t stream)
{
  const float* x         = (const float*)d_in[0];
  const float* wind_dirs = (const float*)d_in[1];
  const float* adj       = (const float*)d_in[2];
  const float* bearings  = (const float*)d_in[3];
  const float* in_W      = (const float*)d_in[4];
  const float* in_b      = (const float*)d_in[5];
  const float* station   = (const float*)d_in[6];
  const float* ln_g      = (const float*)d_in[7];
  const float* ln_b      = (const float*)d_in[8];
  const float* Wq        = (const float*)d_in[9];
  const float* Wk        = (const float*)d_in[10];
  const float* Wv        = (const float*)d_in[11];
  const float* Wo        = (const float*)d_in[12];
  const float* windW     = (const float*)d_in[13];
  const float* windB     = (const float*)d_in[14];
  const float* W1        = (const float*)d_in[15];
  const float* b1        = (const float*)d_in[16];
  const float* W2        = (const float*)d_in[17];
  const float* b2        = (const float*)d_in[18];
  const float* W3        = (const float*)d_in[19];
  const float* b3        = (const float*)d_in[20];
  float* out = (float*)d_out;

  const size_t T = TOK_;
  float* ws  = (float*)d_ws;
  float* h_x = ws;                  // T*896 fp32 residual master
  float* qb  = h_x + T * D_;        // T*896 fp32
  float* kb  = qb  + T * D_;        // T*896 fp32
  float* vb  = kb  + T * D_;        // T*896 fp32
  __bf16* xb   = (__bf16*)(vb + T * D_); // T*64
  __bf16* hlnb = xb   + T * DIN_;   // T*896
  __bf16* obb  = hlnb + T * D_;     // T*896
  __bf16* hxb  = obb  + T * D_;     // T*896
  __bf16* h1b  = hxb  + T * D_;     // T*1792
  __bf16* h2b  = h1b  + T * DHID_;  // T*1792

  #define GEMM(EPI, A, Wm, bi, res, st, Cf, Cb, M, Nn, K)                         \
    gemm_bf16_wmma<EPI><<<dim3((Nn)/64, (M)/128), 128, 0, stream>>>(              \
        A, Wm, bi, res, st, Cf, Cb, M, Nn, K)

  // x -> bf16, then x0 = x @ in_W + in_b + station_emb  (fp32 h_x)
  cvt_kernel<<<(TOK_ * DIN_ + 255) / 256, 256, 0, stream>>>(x, xb, TOK_ * DIN_);
  GEMM(E_BIAS | E_STATION | E_OUTF, xb, in_W, in_b, nullptr, station,
       h_x, nullptr, TOK_, D_, DIN_);

  for (int l = 0; l < L_; ++l) {
    ln_kernel<<<TOK_, 256, 0, stream>>>(h_x, ln_g + (size_t)l * D_,
                                        ln_b + (size_t)l * D_, hlnb);
    GEMM(E_OUTF, hlnb, Wq + (size_t)l * D_ * D_, nullptr, nullptr, nullptr,
         qb, nullptr, TOK_, D_, D_);
    GEMM(E_OUTF, hlnb, Wk + (size_t)l * D_ * D_, nullptr, nullptr, nullptr,
         kb, nullptr, TOK_, D_, D_);
    GEMM(E_OUTF, hlnb, Wv + (size_t)l * D_ * D_, nullptr, nullptr, nullptr,
         vb, nullptr, TOK_, D_, D_);
    int rt = TOK_ * H_ * 32;
    rope_kernel<<<(rt + 255) / 256, 256, 0, stream>>>(qb, kb);
    attn_kernel<<<B_ * H_, 256, 0, stream>>>(qb, kb, vb, adj, bearings, wind_dirs,
                                             windW + (size_t)l * 2 * H_,
                                             windB + (size_t)l * H_, obb);
    // h_x = h_x + o @ Wo   (fp32 master + bf16 shadow for W1)
    GEMM(E_RESID | E_OUTF | E_OUTB, obb, Wo + (size_t)l * D_ * D_, nullptr, h_x,
         nullptr, h_x, hxb, TOK_, D_, D_);
    // FFN (bf16 activations only)
    GEMM(E_BIAS | E_SILU | E_OUTB, hxb, W1 + (size_t)l * D_ * DHID_,
         b1 + (size_t)l * DHID_, nullptr, nullptr, nullptr, h1b, TOK_, DHID_, D_);
    GEMM(E_BIAS | E_SILU | E_OUTB, h1b, W2 + (size_t)l * DHID_ * DHID_,
         b2 + (size_t)l * DHID_, nullptr, nullptr, nullptr, h2b, TOK_, DHID_, DHID_);
    float* dst = (l == L_ - 1) ? out : h_x;
    // h_x = h_x + h2 @ W3 + b3  (fp32 master + bf16 shadow; next layer's Wo GEMM rewrites it before W1)
    GEMM(E_BIAS | E_RESID | E_OUTF | E_OUTB, h2b, W3 + (size_t)l * DHID_ * D_,
         b3 + (size_t)l * D_, h_x, nullptr, dst, hxb, TOK_, D_, DHID_);
  }
  #undef GEMM

  (void)in_sizes; (void)n_in; (void)out_size; (void)ws_size;
}
